// DeformConv2d_7069516169323
// MI455X (gfx1250) — matrix operand order repaired and checked by reading, not compiled, not run on hardware
//
#include <hip/hip_runtime.h>
#include <hip/hip_bf16.h>
#include <math.h>

typedef __attribute__((ext_vector_type(16))) _Float16 v16h;
typedef __attribute__((ext_vector_type(8)))  float    v8f;

#define Nn     8
#define Cc     64
#define Hh     112
#define Ww     112
#define Oo     64
#define TAPS   9
#define KD     576      // C * 9
#define HW     12544    // 112*112
#define P_TOT  100352   // N * HW
#define PTILE  32       // positions per block (divides HW)
#define VROW   592      // padded row length (halves); 592*2 = 1184 B = 37*32 -> 32B aligned rows

// ---------------------------------------------------------------------------
// Kernel 1: convert main conv weight [O][C][3][3] f32 -> f16 [O][KD], k=c*9+t
// ---------------------------------------------------------------------------
__global__ __launch_bounds__(256) void dcn_cvt_w(const float* __restrict__ w,
                                                 _Float16* __restrict__ w16) {
    int i = blockIdx.x * 256 + threadIdx.x;
    if (i < Oo * KD) w16[i] = (_Float16)w[i];
}

// ---------------------------------------------------------------------------
// Kernel 2: offset/mask predictor (3x3 conv, 27 channels) -> py, px, mask
// ---------------------------------------------------------------------------
__global__ __launch_bounds__(256) void dcn_offsets(const float* __restrict__ x,
                                                   const float* __restrict__ ow,
                                                   const float* __restrict__ ob,
                                                   float* __restrict__ py,
                                                   float* __restrict__ px,
                                                   float* __restrict__ msk) {
    __shared__ float wl[27 * KD];     // 62,208 B
    int tid = threadIdx.x;
    for (int i = tid; i < 27 * KD; i += 256) wl[i] = ow[i];
    __syncthreads();

    int p    = blockIdx.x * 256 + tid;       // grid is exactly P_TOT/256
    int nimg = p / HW;
    int rem  = p - nimg * HW;
    int y    = rem / Ww;
    int xx   = rem - y * Ww;

    float acc[27];
#pragma unroll
    for (int i = 0; i < 27; ++i) acc[i] = ob[i];

    const float* xb = x + (size_t)nimg * Cc * HW;
    for (int c = 0; c < Cc; ++c) {
        float xv[9];
#pragma unroll
        for (int ky = 0; ky < 3; ++ky) {
            int  yy  = y + ky - 1;
            bool yok = (yy >= 0) && (yy < Hh);
#pragma unroll
            for (int kx = 0; kx < 3; ++kx) {
                int  xc = xx + kx - 1;
                bool ok = yok && (xc >= 0) && (xc < Ww);
                xv[ky * 3 + kx] = ok ? xb[c * HW + yy * Ww + xc] : 0.0f;
            }
        }
        const float* wr = &wl[c * 9];
#pragma unroll
        for (int oc = 0; oc < 27; ++oc) {
            float a = acc[oc];
#pragma unroll
            for (int i = 0; i < 9; ++i) a = fmaf(xv[i], wr[oc * KD + i], a);
            acc[oc] = a;
        }
    }

#pragma unroll
    for (int t = 0; t < 9; ++t) {
        float base_y = (float)(y  - 1 + t / 3);
        float base_x = (float)(xx - 1 + t % 3);
        int   oidx   = (nimg * 9 + t) * HW + rem;
        py[oidx]  = acc[2 * t]     + base_y;
        px[oidx]  = acc[2 * t + 1] + base_x;
        msk[oidx] = 1.0f / (1.0f + expf(-acc[18 + t]));
    }
}

// ---------------------------------------------------------------------------
// Kernel 3: bilinear gather -> LDS f16 tile -> WMMA implicit GEMM -> output
// Block: 256 thr (8 waves). Tile: 64 (all O) x 32 positions.
// ---------------------------------------------------------------------------
__global__ __launch_bounds__(256) void dcn_main(const float* __restrict__ x,
                                                const _Float16* __restrict__ w16,
                                                const float* __restrict__ bias,
                                                const float* __restrict__ py,
                                                const float* __restrict__ px,
                                                const float* __restrict__ msk,
                                                float* __restrict__ out) {
    __shared__ __align__(32) _Float16 V[PTILE * VROW];   // 37,888 B

    int tid   = threadIdx.x;
    int pbase = blockIdx.x * PTILE;
    int nimg  = pbase / HW;            // PTILE divides HW: one image per block
    int remb  = pbase - nimg * HW;
    const float* xb = x + (size_t)nimg * Cc * HW;

    // ---- stage V[pl][c*9+t] = bilinear(x, py, px) * mask, as f16 ----
    for (int pair = tid; pair < TAPS * PTILE; pair += 256) {
        int t  = pair / PTILE;
        int pl = pair - t * PTILE;
        int cidx = (nimg * 9 + t) * HW + remb + pl;

        float fy = py[cidx], fx = px[cidx], m = msk[cidx];
        float y0f = floorf(fy), x0f = floorf(fx);
        int   y0  = (int)y0f,   x0  = (int)x0f;
        float ay  = fy - y0f,   ax  = fx - x0f;

        float w00 = (1.0f - ay) * (1.0f - ax);
        float w01 = (1.0f - ay) * ax;
        float w10 = ay * (1.0f - ax);
        float w11 = ay * ax;

        bool y0ok = (y0 >= 0) && (y0 <= Hh - 1);
        bool y1ok = (y0 + 1 >= 0) && (y0 + 1 <= Hh - 1);
        bool x0ok = (x0 >= 0) && (x0 <= Ww - 1);
        bool x1ok = (x0 + 1 >= 0) && (x0 + 1 <= Ww - 1);
        w00 = (y0ok && x0ok) ? w00 * m : 0.0f;   // fold modulation mask in
        w01 = (y0ok && x1ok) ? w01 * m : 0.0f;
        w10 = (y1ok && x0ok) ? w10 * m : 0.0f;
        w11 = (y1ok && x1ok) ? w11 * m : 0.0f;

        int yc0 = min(max(y0, 0), Hh - 1),     yc1 = min(max(y0 + 1, 0), Hh - 1);
        int xc0 = min(max(x0, 0), Ww - 1),     xc1 = min(max(x0 + 1, 0), Ww - 1);
        int i00 = yc0 * Ww + xc0, i01 = yc0 * Ww + xc1;
        int i10 = yc1 * Ww + xc0, i11 = yc1 * Ww + xc1;

        _Float16* vrow = &V[pl * VROW + t];
#pragma unroll 4
        for (int c = 0; c < Cc; ++c) {
            const float* xc = xb + c * HW;
            float v = w00 * xc[i00] + w01 * xc[i01] + w10 * xc[i10] + w11 * xc[i11];
            vrow[c * 9] = (_Float16)v;
        }
    }
    __syncthreads();

    // ---- GEMM: wave wv -> (mi = wv>>1) x (psub = wv&1) 16x16 tile ----
    int wv   = tid >> 5;
    int lane = tid & 31;
    int mi   = wv >> 1;
    int psub = wv & 1;
    int half = lane >> 4;
    int l16  = lane & 15;

    v8f acc = {};
    const _Float16* arow = w16 + (mi * 16 + l16) * KD;        // A: W[m][k], 16-bit A layout
    const _Float16* brow = &V[(psub * 16 + l16) * VROW];      // B: V[p][k], 32x16 B layout

#pragma unroll
    for (int k0 = 0; k0 < KD; k0 += 32) {
        v16h a, b;
        *((uint4*)&a)     = *(const uint4*)(arow + k0 + 8 * half);        // K = k0 + 8h .. +7
        *((uint4*)&a + 1) = *(const uint4*)(arow + k0 + 16 + 8 * half);   // K = k0+16+8h .. +7
        *((uint4*)&b)     = *(const uint4*)(brow + k0 + 8 * half);        // K = k0 + 8h .. +7
        *((uint4*)&b + 1) = *(const uint4*)(brow + k0 + 16 + 8 * half);   // K = k0+16+8h .. +7
        acc = __builtin_amdgcn_wmma_f32_16x16x32_f16(
            /*neg_a=*/false, a, /*neg_b=*/false, b,
            /*c_mod=*/(short)0, acc, /*reuse_a=*/false, /*reuse_b=*/false);
    }

    // ---- store: f32 C/D layout (VGPR r -> M = mi*16 + 8*half + r, N = l16) ----
    int o_base = mi * 16 + half * 8;
    int pl     = psub * 16 + l16;
    int rem    = remb + pl;
    float* op  = out + (size_t)nimg * Oo * HW + rem;
#pragma unroll
    for (int r = 0; r < 8; ++r) {
        int o = o_base + r;
        op[(size_t)o * HW] = acc[r] + bias[o];
    }
}

// ---------------------------------------------------------------------------
extern "C" void kernel_launch(void* const* d_in, const int* in_sizes, int n_in,
                              void* d_out, int out_size, void* d_ws, size_t ws_size,
                              hipStream_t stream) {
    const float* x      = (const float*)d_in[0];
    const float* weight = (const float*)d_in[1];
    const float* bias   = (const float*)d_in[2];
    const float* ow     = (const float*)d_in[3];
    const float* ob     = (const float*)d_in[4];
    float* out = (float*)d_out;

    const size_t planes = (size_t)Nn * TAPS * HW;   // 903,168 f32 each
    float*     py  = (float*)d_ws;
    float*     px  = py + planes;
    float*     mk  = px + planes;
    _Float16*  w16 = (_Float16*)(mk + planes);      // 73,728 B, 16B aligned

    dcn_cvt_w  <<<(Oo * KD + 255) / 256, 256, 0, stream>>>(weight, w16);
    dcn_offsets<<<P_TOT / 256,          256, 0, stream>>>(x, ow, ob, py, px, mk);
    dcn_main   <<<P_TOT / PTILE,        256, 0, stream>>>(x, w16, bias, py, px, mk, out);
}
